// AFF_1116691497756
// MI455X (gfx1250) — matrix operand order repaired and checked by reading, not compiled, not run on hardware
//
#include <hip/hip_runtime.h>
#include <hip/hip_bf16.h>
#include <math.h>

// ---------- CDNA5 WMMA types ----------
typedef __bf16 bf16;
typedef bf16  v8bf  __attribute__((ext_vector_type(8)));
typedef bf16  v16bf __attribute__((ext_vector_type(16)));
typedef float v8f   __attribute__((ext_vector_type(8)));

#define EPSBN 1e-5f
#define B_   32
#define C_   256
#define I_   64
#define L_   4096
#define NT   64           // positions per block tile

__device__ __forceinline__ float sigm(float x) { return 1.0f / (1.0f + __expf(-x)); }

// ============================================================
// Fold BN into 1x1-conv weights for the heavy local branches
// (branch 0 -> slot 0, branch 2 -> slot 1), convert to bf16.
// ============================================================
__global__ void k_fold(const float* w1, const float* b1,
                       const float* g1, const float* be1, const float* m1, const float* v1,
                       const float* w2, const float* b2,
                       const float* g2, const float* be2, const float* m2, const float* v2,
                       bf16* w1f, float* b1f, bf16* w2f, float* b2f)
{
    int idx = blockIdx.x * blockDim.x + threadIdx.x;
    if (idx >= 2 * I_ * C_) return;
    int slot = idx / (I_ * C_);       // 0 or 1
    int br   = slot * 2;              // branch 0 or 2
    int rem  = idx % (I_ * C_);

    { // conv1: [I][C], scale rows by bn1 scale
        int m = rem / C_, c = rem % C_;
        float s = g1[br * I_ + m] * rsqrtf(v1[br * I_ + m] + EPSBN);
        w1f[idx] = (bf16)(w1[br * I_ * C_ + m * C_ + c] * s);
        if (c == 0)
            b1f[slot * I_ + m] = b1[br * I_ + m] * s
                               + (be1[br * I_ + m] - m1[br * I_ + m] * s);
    }
    { // conv2: [C][I], scale rows by bn2 scale
        int co = rem / I_, i = rem % I_;
        float s = g2[br * C_ + co] * rsqrtf(v2[br * C_ + co] + EPSBN);
        w2f[idx] = (bf16)(w2[br * C_ * I_ + co * I_ + i] * s);
        if (i == 0)
            b2f[slot * C_ + co] = b2[br * C_ + co] * s
                                + (be2[br * C_ + co] - m2[br * C_ + co] * s);
    }
}

// ============================================================
// Per-(b,c) sums over L of (x_a + x_b) and of x_c.
// ============================================================
__global__ void k_sums(const float* xa, const float* xb, const float* xc,
                       float* sum_ab, float* sum_c)
{
    int bc = blockIdx.x;                       // 0 .. B*C-1
    const size_t base = (size_t)bc * L_;
    float sab = 0.f, sc = 0.f;
    for (int l = threadIdx.x; l < L_; l += 256) {
        sab += xa[base + l] + xb[base + l];
        sc  += xc[base + l];
    }
    __shared__ float r0[256], r1[256];
    r0[threadIdx.x] = sab; r1[threadIdx.x] = sc;
    __syncthreads();
    for (int s = 128; s > 0; s >>= 1) {
        if (threadIdx.x < s) {
            r0[threadIdx.x] += r0[threadIdx.x + s];
            r1[threadIdx.x] += r1[threadIdx.x + s];
        }
        __syncthreads();
    }
    if (threadIdx.x == 0) { sum_ab[bc] = r0[0]; sum_c[bc] = r1[0]; }
}

// ============================================================
// Global-pool attention branch (tiny f32 per-batch MLP).
// mean = (s0 [+ s1]) * invL ; g[b,c] = bn2(W2 relu(bn1(W1 mean + b1)) + b2)
// ============================================================
__global__ void k_gatt(int br, const float* s0, const float* s1, float invL,
                       const float* w1, const float* b1,
                       const float* g1, const float* be1, const float* m1, const float* v1,
                       const float* w2, const float* b2,
                       const float* g2, const float* be2, const float* m2, const float* v2,
                       float* gout)
{
    int b = blockIdx.x, tid = threadIdx.x;
    __shared__ float mS[C_], hS[I_];
    float mv = s0[b * C_ + tid];
    if (s1) mv += s1[b * C_ + tid];
    mS[tid] = mv * invL;
    __syncthreads();
    if (tid < I_) {
        float acc = 0.f;
        const float* wr = w1 + br * I_ * C_ + tid * C_;
        for (int c = 0; c < C_; ++c) acc += wr[c] * mS[c];
        float s = g1[br * I_ + tid] * rsqrtf(v1[br * I_ + tid] + EPSBN);
        float y = (acc + b1[br * I_ + tid]) * s
                + (be1[br * I_ + tid] - m1[br * I_ + tid] * s);
        hS[tid] = fmaxf(y, 0.f);
    }
    __syncthreads();
    {
        float acc = 0.f;
        const float* wr = w2 + br * C_ * I_ + tid * I_;
        for (int i = 0; i < I_; ++i) acc += wr[i] * hS[i];
        float s = g2[br * C_ + tid] * rsqrtf(v2[br * C_ + tid] + EPSBN);
        gout[b * C_ + tid] = (acc + b2[br * C_ + tid]) * s
                           + (be2[br * C_ + tid] - m2[br * C_ + tid] * s);
    }
}

// ============================================================
// Main fused kernel: per [NT position x 256 channel] tile:
//   GEMM1 (64x NT, K=256) -> ReLU -> GEMM2 (256 x NT, K=64)
//   -> sigmoid(+global term) -> blend -> store.
// MODE 1: x = x_a + x_b ; xo1 = xb + w*(xa-xb) ; store xo1, wei, Σxo1
// MODE 2: x = xo1 + x_c ; out = 0.5*(xo1*(1+wei2) + xc*(1-wei))
// ============================================================
template<int MODE>
__global__ __launch_bounds__(256) void k_main(
    const float* __restrict__ pA, const float* __restrict__ pB,
    const float* __restrict__ pWei,
    const bf16* __restrict__ w1f, const float* __restrict__ b1f,
    const bf16* __restrict__ w2f, const float* __restrict__ b2f,
    const float* __restrict__ gvec,
    float* __restrict__ out0, float* __restrict__ out1,
    float* __restrict__ sumAcc)
{
    const int tilesPerB = L_ / NT;                 // 64
    const int b   = blockIdx.x / tilesPerB;
    const int l0  = (blockIdx.x % tilesPerB) * NT;
    const int tid = threadIdx.x;
    const int lane = tid & 31, wv = tid >> 5;      // 8 waves of 32
    const int nl = lane & 15, half = lane >> 4;

    __shared__ __align__(32) bf16 Xs[NT][C_];      // activations, position-major: [n][k]
    __shared__ __align__(32) bf16 Hs[NT][I_];      // hidden, position-major: [n][k]
    __shared__ float sred[C_];

    if (MODE == 1) sred[tid] = 0.f;

    const size_t base = ((size_t)b * C_) * L_ + l0;

    // Stage x = pA + pB into LDS as bf16 (coalesced along l)
    for (int idx = tid; idx < C_ * NT; idx += 256) {
        int c = idx >> 6, n = idx & (NT - 1);
        size_t g = base + (size_t)c * L_ + n;
        Xs[n][c] = (bf16)(pA[g] + pB[g]);
    }
    __syncthreads();

    // ---------------- GEMM1: H = relu(W1' X + b1') ----------------
    // 4 m-tiles x 4 n-tiles = 16 tiles; 2 per wave; K = 256 -> 8 wmma steps
    for (int t = 2 * wv; t < 2 * wv + 2; ++t) {
        int mt = t >> 2, nt = t & 3;
        v8f acc = {};
        for (int kk = 0; kk < 8; ++kk) {
            int k0 = kk * 32;
            union { v16bf v; v8bf h[2]; } ua;       // A: 16x32, row-major weights
            const bf16* pa = w1f + (mt * 16 + nl) * C_ + k0 + 16 * half;
            ua.h[0] = *(const v8bf*)pa;             // K = k0+16h   .. +7
            ua.h[1] = *(const v8bf*)(pa + 8);       // K = k0+16h+8 .. +7
            v16bf bfr = *(const v16bf*)(&Xs[nt * 16 + nl][k0 + 16 * half]); // B: 32x16
            acc = __builtin_amdgcn_wmma_f32_16x16x32_bf16(
                      false, ua.v, false, bfr, (short)0, acc, false, false);
        }
        for (int r = 0; r < 8; ++r) {               // C layout: m = r+8*half, n = nl
            int m = mt * 16 + r + 8 * half;
            int n = nt * 16 + nl;
            Hs[n][m] = (bf16)fmaxf(acc[r] + b1f[m], 0.f);
        }
    }
    __syncthreads();

    // ---------------- GEMM2 + epilogue ----------------
    // 16 m-tiles x 4 n-tiles = 64 tiles; 8 per wave; K = 64 -> 2 wmma steps
    for (int j = 0; j < 8; ++j) {
        int t = wv * 8 + j;
        int mt = t >> 2, nt = t & 3;
        v8f acc = {};
        for (int kk = 0; kk < 2; ++kk) {
            int k0 = kk * 32;
            union { v16bf v; v8bf h[2]; } ua;
            const bf16* pa = w2f + (mt * 16 + nl) * I_ + k0 + 16 * half;
            ua.h[0] = *(const v8bf*)pa;
            ua.h[1] = *(const v8bf*)(pa + 8);
            v16bf bfr = *(const v16bf*)(&Hs[nt * 16 + nl][k0 + 16 * half]);
            acc = __builtin_amdgcn_wmma_f32_16x16x32_bf16(
                      false, ua.v, false, bfr, (short)0, acc, false, false);
        }
        for (int r = 0; r < 8; ++r) {
            int c = mt * 16 + r + 8 * half;
            int l = nt * 16 + nl;
            size_t g = base + (size_t)c * L_ + l;
            float w = sigm(acc[r] + b2f[c] + gvec[b * C_ + c]);
            if (MODE == 1) {
                float xa = pA[g], xb = pB[g];
                float xo1 = xb + w * (xa - xb);     // xa*w + xb*(1-w)
                out0[g] = xo1;
                out1[g] = w;
                atomicAdd(&sred[c], xo1);           // per-channel partial sum
            } else {
                float xo1 = pA[g], xc = pB[g], wOld = pWei[g];
                out0[g] = 0.5f * (xo1 * (1.f + w) + xc * (1.f - wOld));
            }
        }
    }
    if (MODE == 1) {
        __syncthreads();
        atomicAdd(&sumAcc[b * C_ + tid], sred[tid]);
    }
}

// ============================================================
extern "C" void kernel_launch(void* const* d_in, const int* in_sizes, int n_in,
                              void* d_out, int out_size, void* d_ws, size_t ws_size,
                              hipStream_t stream)
{
    const float* xa  = (const float*)d_in[0];
    const float* xb  = (const float*)d_in[1];
    const float* xc  = (const float*)d_in[2];
    const float* w1  = (const float*)d_in[3];
    const float* b1  = (const float*)d_in[4];
    const float* g1  = (const float*)d_in[5];
    const float* be1 = (const float*)d_in[6];
    const float* m1  = (const float*)d_in[7];
    const float* v1  = (const float*)d_in[8];
    const float* w2  = (const float*)d_in[9];
    const float* b2  = (const float*)d_in[10];
    const float* g2  = (const float*)d_in[11];
    const float* be2 = (const float*)d_in[12];
    const float* m2  = (const float*)d_in[13];
    const float* v2  = (const float*)d_in[14];

    char* ws = (char*)d_ws;
    const size_t NELT = (size_t)B_ * C_ * L_;       // 33,554,432
    float* xo1     = (float*)(ws);                              // 128 MB
    float* weiBuf  = (float*)(ws + NELT * 4);                   // 128 MB
    char*  sm      = ws + 2 * NELT * 4;
    float* sum_ab  = (float*)(sm);                              // B*C
    float* sum_c   = (float*)(sm + 32768);
    float* sum_xo1 = (float*)(sm + 65536);
    float* gv1     = (float*)(sm + 98304);
    float* gv2     = (float*)(sm + 131072);
    float* b1f     = (float*)(sm + 163840);                     // 2*64
    float* b2f     = (float*)(sm + 164352);                     // 2*256
    bf16*  w1f     = (bf16*)(sm + 166400);                      // 2*64*256 bf16
    bf16*  w2f     = (bf16*)(sm + 231936);                      // 2*256*64 bf16

    const float invL = 1.0f / (float)L_;

    hipMemsetAsync(sum_xo1, 0, B_ * C_ * sizeof(float), stream);

    k_fold<<<(2 * I_ * C_ + 255) / 256, 256, 0, stream>>>(
        w1, b1, g1, be1, m1, v1, w2, b2, g2, be2, m2, v2, w1f, b1f, w2f, b2f);

    k_sums<<<B_ * C_, 256, 0, stream>>>(xa, xb, xc, sum_ab, sum_c);

    k_gatt<<<B_, 256, 0, stream>>>(1, sum_ab, nullptr, invL,
        w1, b1, g1, be1, m1, v1, w2, b2, g2, be2, m2, v2, gv1);

    const int grid = B_ * (L_ / NT);                            // 2048
    k_main<1><<<grid, 256, 0, stream>>>(
        xa, xb, nullptr, w1f, b1f, w2f, b2f, gv1, xo1, weiBuf, sum_xo1);

    k_gatt<<<B_, 256, 0, stream>>>(3, sum_xo1, sum_c, invL,
        w1, b1, g1, be1, m1, v1, w2, b2, g2, be2, m2, v2, gv2);

    k_main<2><<<grid, 256, 0, stream>>>(
        xo1, xc, weiBuf,
        w1f + I_ * C_, b1f + I_, w2f + C_ * I_, b2f + C_,
        gv2, (float*)d_out, nullptr, nullptr);
}
